// SequenceBackbone_68058051772875
// MI455X (gfx1250) — compile-verified, matrix operand order checked by reading, not run on hardware
//
#include <hip/hip_runtime.h>
#include <cstdint>
#include <cstddef>

// ---------------------------------------------------------------------------
// Mamba backbone for MI455X (gfx1250, wave32, WMMA + async global->LDS DMA).
// GEMMs: bf16 operands (pre-converted), f32 accumulation via
// v_wmma_f32_16x16x32_bf16; K-panels double-buffered in LDS and filled with
// GLOBAL_LOAD_ASYNC_TO_LDS_B128 (ASYNCcnt), overlapping DMA with WMMA.
// Wave tile 32x64 (2x4 WMMA): 12 ds_load_b128 feed 8 WMMAs per K-step.
// ---------------------------------------------------------------------------

typedef __attribute__((ext_vector_type(16))) __bf16       v16bf;
typedef __attribute__((ext_vector_type(8)))  float        v8f;
typedef __attribute__((ext_vector_type(4)))  __bf16       bf4;
typedef __attribute__((ext_vector_type(4)))  unsigned int uv4;

#define D_MODEL 512
#define D_INNER 1024
#define D_STATE 16
#define D_CONV  4
#define DT_RANK 32
#define BATCH   4
#define SEQ     2048
#define NTOK    (BATCH * SEQ)      /* 8192 */
#define NLAYERS 3
#define DBC_W   (DT_RANK + 2 * D_STATE)   /* 64 */

#define BM 256
#define BN 64
#define BK 32

// ---------------------------------------------------------------------------
// Async 16-byte global -> LDS copy (gfx1250 GLOBAL_LOAD_ASYNC_TO_LDS_B128).
// LDS operand = low 32 bits of the generic pointer (the LDS aperture maps
// addr[31:0] to the LDS offset). Tracked by ASYNCcnt; in-order completion.
// ---------------------------------------------------------------------------
__device__ __forceinline__ void async_copy16(const __bf16* gsrc, __bf16* ldst)
{
    const unsigned      loff  = (unsigned)(uintptr_t)ldst;
    const unsigned long gaddr = (unsigned long)(uintptr_t)gsrc;
    asm volatile("global_load_async_to_lds_b128 %0, %1, off"
                 :: "v"(loff), "v"(gaddr) : "memory");
}

// ---------------------------------------------------------------------------
// Generic NT GEMM: C[M x N] = A[M x K] * B[N x K]^T, A/B bf16, C fp32
// (+ optional mirrored bf16 copy of C for chained GEMMs).
// Block = 256 threads = 8 waves, each wave owns a 32x64 sub-tile
// (2x4 WMMA 16x16 accumulators). M%256 == N%64 == K%32 == 0 everywhere.
// ---------------------------------------------------------------------------
__device__ __forceinline__ void stage_async(
    const __bf16* __restrict__ A, int lda,
    const __bf16* __restrict__ Bm, int ldb,
    int m0, int n0, int k0,
    __bf16 (*sA)[BK], __bf16 (*sB)[BK], int tid)
{
    // A tile: 256x32 bf16 = 1024 x 16B chunks -> 4 per thread
#pragma unroll
    for (int p = 0; p < 4; ++p) {
        const int c   = tid + (p << 8);
        const int row = c >> 2;
        const int kq  = (c & 3) << 3;
        async_copy16(A + (size_t)(m0 + row) * lda + k0 + kq, &sA[row][kq]);
    }
    // B tile: 64x32 bf16 = 256 x 16B chunks -> 1 per thread
    {
        const int row = tid >> 2;
        const int kq  = (tid & 3) << 3;
        async_copy16(Bm + (size_t)(n0 + row) * ldb + k0 + kq, &sB[row][kq]);
    }
}

__global__ __launch_bounds__(256) void gemm_nt_wmma(
    const __bf16* __restrict__ A, int lda,
    const __bf16* __restrict__ Bm, int ldb,
    float* __restrict__ C, __bf16* __restrict__ Cbf, int ldc, int K)
{
    __shared__ __bf16 sA[2][BM][BK];   // 32 KB (double-buffered)
    __shared__ __bf16 sB[2][BN][BK];   //  8 KB

    const int tid  = threadIdx.x;
    const int lane = tid & 31;
    const int wave = tid >> 5;
    const int m0   = blockIdx.y * BM;
    const int n0   = blockIdx.x * BN;
    const int wm   = wave * 32;          // 8 waves stacked along M

    v8f acc[2][4];
#pragma unroll
    for (int i = 0; i < 2; ++i)
#pragma unroll
        for (int j = 0; j < 4; ++j)
#pragma unroll
            for (int r = 0; r < 8; ++r) acc[i][j][r] = 0.0f;

    const int KT = K / BK;
    stage_async(A, lda, Bm, ldb, m0, n0, 0, sA[0], sB[0], tid);

    for (int kt = 0; kt < KT; ++kt) {
        const int cur = kt & 1;

        // Issue DMA for the next K panel, then wait only for the current one
        // (5 async ops per thread per stage; in-order completion).
        if (kt + 1 < KT) {
            stage_async(A, lda, Bm, ldb, m0, n0, (kt + 1) * BK,
                        sA[1 - cur], sB[1 - cur], tid);
            asm volatile("s_wait_asynccnt 0x5" ::: "memory");
        } else {
            asm volatile("s_wait_asynccnt 0x0" ::: "memory");
        }
        __syncthreads();

        if (kt + 2 < KT) {   // keep L2 warm for the panel after next
            __builtin_prefetch(A  + (size_t)(m0 + tid) * lda + (kt + 2) * BK, 0, 1);
            __builtin_prefetch(Bm + (size_t)(n0 + (tid >> 2)) * ldb + (kt + 2) * BK, 0, 1);
        }

        // ---- fragment assembly per ISA 16-bit A/B layout ------------------
        // lane<16 : row=lane,    K = {0..7, 16..23}
        // lane>=16: row=lane-16, K = {8..15, 24..31}
        const int fr   = lane & 15;
        const int koff = (lane >> 4) << 3;
        v16bf afrag[2], bfrag[4];
#pragma unroll
        for (int i = 0; i < 2; ++i) {
            const __bf16* pa = &sA[cur][wm + i * 16 + fr][koff];
            ((uv4*)&afrag[i])[0] = *(const uv4*)pa;
            ((uv4*)&afrag[i])[1] = *(const uv4*)(pa + 16);
        }
#pragma unroll
        for (int j = 0; j < 4; ++j) {
            const __bf16* pb = &sB[cur][j * 16 + fr][koff];
            ((uv4*)&bfrag[j])[0] = *(const uv4*)pb;
            ((uv4*)&bfrag[j])[1] = *(const uv4*)(pb + 16);
        }

#pragma unroll
        for (int i = 0; i < 2; ++i)
#pragma unroll
            for (int j = 0; j < 4; ++j)
                acc[i][j] = __builtin_amdgcn_wmma_f32_16x16x32_bf16(
                    false, afrag[i], false, bfrag[j],
                    (short)0, acc[i][j], false, false);
        __syncthreads();   // reads of buf `cur` done before it is re-filled
    }

    // ---- store: VGPR r -> M = r (+8 for upper lane half), N = lane%16 -----
    const int fr = lane & 15;
    const int ms = (lane >> 4) << 3;
#pragma unroll
    for (int i = 0; i < 2; ++i)
#pragma unroll
        for (int j = 0; j < 4; ++j) {
            const size_t rbase = (size_t)(m0 + wm + i * 16 + ms);
            const int    col   = n0 + j * 16 + fr;
            float* cp = C + rbase * ldc + col;
#pragma unroll
            for (int r = 0; r < 8; ++r)
                cp[(size_t)r * ldc] = acc[i][j][r];
            if (Cbf) {
                __bf16* cb = Cbf + rbase * ldc + col;
#pragma unroll
                for (int r = 0; r < 8; ++r)
                    cb[(size_t)r * ldc] = (__bf16)acc[i][j][r];
            }
        }
}

// ---------------------------------------------------------------------------
// fp32 -> bf16 bulk conversion (weights; 8 elements per thread).
// ---------------------------------------------------------------------------
__global__ void f32_to_bf16_kernel(const float* __restrict__ src,
                                   __bf16* __restrict__ dst)
{
    const size_t i = ((size_t)blockIdx.x * blockDim.x + threadIdx.x) * 8;
    const float4 a = *(const float4*)(src + i);
    const float4 b = *(const float4*)(src + i + 4);
    bf4 t0; t0[0] = (__bf16)a.x; t0[1] = (__bf16)a.y; t0[2] = (__bf16)a.z; t0[3] = (__bf16)a.w;
    bf4 t1; t1[0] = (__bf16)b.x; t1[1] = (__bf16)b.y; t1[2] = (__bf16)b.z; t1[3] = (__bf16)b.w;
    *(bf4*)(dst + i)     = t0;
    *(bf4*)(dst + i + 4) = t1;
}

// ---------------------------------------------------------------------------
// Depthwise causal conv(4) + bias + SiLU; emits fp32 (for scan) and bf16
// (for the x_proj GEMM operand).
// ---------------------------------------------------------------------------
__global__ void conv_silu_kernel(const float* __restrict__ xz,
                                 const float* __restrict__ cw,
                                 const float* __restrict__ cb,
                                 float* __restrict__ xc,
                                 __bf16* __restrict__ xcbf)
{
    const int i = blockIdx.x * blockDim.x + threadIdx.x;   // < NTOK*D_INNER
    const int d = i & (D_INNER - 1);
    const int m = i >> 10;
    const int b = m >> 11;
    const int n = m & (SEQ - 1);
    float acc = cb[d];
#pragma unroll
    for (int j = 0; j < D_CONV; ++j) {
        const int nn = n - (D_CONV - 1) + j;
        if (nn >= 0)
            acc += xz[(size_t)(b * SEQ + nn) * (2 * D_INNER) + d] * cw[d * D_CONV + j];
    }
    const float s = acc / (1.0f + __expf(-acc));
    xc[(size_t)m * D_INNER + d]   = s;
    xcbf[(size_t)m * D_INNER + d] = (__bf16)s;
}

// ---------------------------------------------------------------------------
// delta = softplus(delta + dt_proj_b)
// ---------------------------------------------------------------------------
__global__ void softplus_bias_kernel(float* __restrict__ delta,
                                     const float* __restrict__ dtb)
{
    const int i = blockIdx.x * blockDim.x + threadIdx.x;
    const int d = i & (D_INNER - 1);
    const float v = delta[i] + dtb[d];
    delta[i] = (v > 20.0f) ? v : log1pf(__expf(v));
}

// ---------------------------------------------------------------------------
// Selective scan (sequential over N) fused with D*u skip and SiLU(z) gating;
// writes the gated output directly as bf16 (out_proj GEMM operand).
// ---------------------------------------------------------------------------
__global__ __launch_bounds__(256) void scan_kernel(
    const float* __restrict__ delta, const float* __restrict__ dbc,
    const float* __restrict__ xc,    const float* __restrict__ xz,
    const float* __restrict__ A_log, const float* __restrict__ Dp,
    __bf16* __restrict__ ybf)
{
    const int tid = threadIdx.x;
    const int b   = blockIdx.x >> 2;                    // 4 blocks per batch
    const int d   = ((blockIdx.x & 3) << 8) + tid;

    float Arow[D_STATE], h[D_STATE];
#pragma unroll
    for (int s = 0; s < D_STATE; ++s) {
        Arow[s] = -__expf(A_log[d * D_STATE + s]);
        h[s]    = 0.0f;
    }
    const float Dd = Dp[d];

    __shared__ float sB[D_STATE], sC[D_STATE];
    for (int n = 0; n < SEQ; ++n) {
        const size_t row = (size_t)b * SEQ + n;
        __syncthreads();
        if (tid < 2 * D_STATE) {
            const float v = dbc[row * DBC_W + DT_RANK + tid];
            if (tid < D_STATE) sB[tid] = v; else sC[tid - D_STATE] = v;
        }
        __syncthreads();
        const float dl = delta[row * D_INNER + d];
        const float u  = xc[row * D_INNER + d];
        const float du = dl * u;
        float yt = 0.0f;
#pragma unroll
        for (int s = 0; s < D_STATE; ++s) {
            h[s] = __expf(dl * Arow[s]) * h[s] + du * sB[s];
            yt  += h[s] * sC[s];
        }
        const float zz = xz[row * (2 * D_INNER) + D_INNER + d];
        const float g  = zz / (1.0f + __expf(-zz));
        ybf[row * D_INNER + d] = (__bf16)((yt + Dd * u) * g);
    }
}

// ---------------------------------------------------------------------------
// h = x * mask  (input masking), fp32 + bf16 outputs.
// ---------------------------------------------------------------------------
__global__ void mask_in_kernel(const float* __restrict__ x,
                               const unsigned char* __restrict__ mask,
                               float* __restrict__ h,
                               __bf16* __restrict__ hbf)
{
    const int i = blockIdx.x * blockDim.x + threadIdx.x;   // < NTOK*D_MODEL
    const int m = i / D_MODEL;
    const float v = mask[m] ? x[i] : 0.0f;
    h[i]   = v;
    hbf[i] = (__bf16)v;
}

// ---------------------------------------------------------------------------
// Final LayerNorm over D_MODEL + output masking. One 128-thread block per row.
// ---------------------------------------------------------------------------
__global__ __launch_bounds__(128) void layernorm_mask_kernel(
    const float* __restrict__ h, const float* __restrict__ nw,
    const float* __restrict__ nb, const unsigned char* __restrict__ mask,
    float* __restrict__ out)
{
    const int row = blockIdx.x;
    const int tid = threadIdx.x;
    const int c   = tid * 4;
    const float4 v = *(const float4*)(h + (size_t)row * D_MODEL + c);

    __shared__ float red[128];
    red[tid] = v.x + v.y + v.z + v.w;
    __syncthreads();
    for (int o = 64; o > 0; o >>= 1) { if (tid < o) red[tid] += red[tid + o]; __syncthreads(); }
    const float mu = red[0] * (1.0f / D_MODEL);
    __syncthreads();

    const float d0 = v.x - mu, d1 = v.y - mu, d2 = v.z - mu, d3 = v.w - mu;
    red[tid] = d0 * d0 + d1 * d1 + d2 * d2 + d3 * d3;
    __syncthreads();
    for (int o = 64; o > 0; o >>= 1) { if (tid < o) red[tid] += red[tid + o]; __syncthreads(); }
    const float rstd = rsqrtf(red[0] * (1.0f / D_MODEL) + 1e-5f);

    const float mv = mask[row] ? 1.0f : 0.0f;
    float4 o4;
    o4.x = (d0 * rstd * nw[c + 0] + nb[c + 0]) * mv;
    o4.y = (d1 * rstd * nw[c + 1] + nb[c + 1]) * mv;
    o4.z = (d2 * rstd * nw[c + 2] + nb[c + 2]) * mv;
    o4.w = (d3 * rstd * nw[c + 3] + nb[c + 3]) * mv;
    *(float4*)(out + (size_t)row * D_MODEL + c) = o4;
}

// ---------------------------------------------------------------------------
// Host-side orchestration (graph-capture safe: only kernel launches).
// ---------------------------------------------------------------------------
extern "C" void kernel_launch(void* const* d_in, const int* /*in_sizes*/, int /*n_in*/,
                              void* d_out, int /*out_size*/, void* d_ws, size_t /*ws_size*/,
                              hipStream_t stream)
{
    const float*         x     = (const float*)d_in[0];
    const unsigned char* mask  = (const unsigned char*)d_in[1];
    const float*         ipw   = (const float*)d_in[2];
    const float*         cw    = (const float*)d_in[3];
    const float*         cb    = (const float*)d_in[4];
    const float*         xpw   = (const float*)d_in[5];
    const float*         dtw   = (const float*)d_in[6];
    const float*         dtb   = (const float*)d_in[7];
    const float*         A_log = (const float*)d_in[8];
    const float*         Dp    = (const float*)d_in[9];
    const float*         opw   = (const float*)d_in[10];
    const float*         nw    = (const float*)d_in[11];
    const float*         nb    = (const float*)d_in[12];

    // ---- workspace carve-up: fp32 region then bf16 region -----------------
    float* wsf = (float*)d_ws;
    size_t off = 0;
    float* h     = wsf + off;  off += (size_t)NTOK * D_MODEL;
    float* xz    = wsf + off;  off += (size_t)NTOK * 2 * D_INNER;
    float* xc    = wsf + off;  off += (size_t)NTOK * D_INNER;
    float* dbc   = wsf + off;  off += (size_t)NTOK * DBC_W;
    float* delta = wsf + off;  off += (size_t)NTOK * D_INNER;

    __bf16* wsb = (__bf16*)(wsf + off);
    size_t bo = 0;
    __bf16* hbf    = wsb + bo;  bo += (size_t)NTOK * D_MODEL;
    __bf16* xcbf   = wsb + bo;  bo += (size_t)NTOK * D_INNER;
    __bf16* dbcbf  = wsb + bo;  bo += (size_t)NTOK * DBC_W;
    __bf16* ybf    = wsb + bo;  bo += (size_t)NTOK * D_INNER;
    __bf16* ipwbf  = wsb + bo;  bo += (size_t)NLAYERS * 2 * D_INNER * D_MODEL;
    __bf16* xpwbf  = wsb + bo;  bo += (size_t)NLAYERS * DBC_W * D_INNER;
    __bf16* dtwbf  = wsb + bo;  bo += (size_t)NLAYERS * D_INNER * DT_RANK;
    __bf16* opwbf  = wsb + bo;  bo += (size_t)NLAYERS * D_MODEL * D_INNER;

    // ---- one-time weight conversion to bf16 (all layers at once) ----------
    f32_to_bf16_kernel<<<(NLAYERS * 2 * D_INNER * D_MODEL) / (8 * 256), 256, 0, stream>>>(ipw, ipwbf);
    f32_to_bf16_kernel<<<(NLAYERS * DBC_W * D_INNER)       / (8 * 256), 256, 0, stream>>>(xpw, xpwbf);
    f32_to_bf16_kernel<<<(NLAYERS * D_INNER * DT_RANK)     / (8 * 256), 256, 0, stream>>>(dtw, dtwbf);
    f32_to_bf16_kernel<<<(NLAYERS * D_MODEL * D_INNER)     / (8 * 256), 256, 0, stream>>>(opw, opwbf);

    mask_in_kernel<<<(NTOK * D_MODEL) / 256, 256, 0, stream>>>(x, mask, h, hbf);

    for (int l = 0; l < NLAYERS; ++l) {
        // xz = h @ in_proj_w^T            (8192 x 2048, K=512)
        gemm_nt_wmma<<<dim3((2 * D_INNER) / BN, NTOK / BM), 256, 0, stream>>>(
            hbf, D_MODEL, ipwbf + (size_t)l * 2 * D_INNER * D_MODEL, D_MODEL,
            xz, (__bf16*)nullptr, 2 * D_INNER, D_MODEL);

        // xc = silu(depthwise_conv(xz[:, :D_INNER]) + conv_b)  [+ bf16 copy]
        conv_silu_kernel<<<(NTOK * D_INNER) / 256, 256, 0, stream>>>(
            xz, cw + (size_t)l * D_INNER * D_CONV, cb + (size_t)l * D_INNER, xc, xcbf);

        // dbc = xc @ x_proj_w^T           (8192 x 64, K=1024)  [+ bf16 copy]
        gemm_nt_wmma<<<dim3(DBC_W / BN, NTOK / BM), 256, 0, stream>>>(
            xcbf, D_INNER, xpwbf + (size_t)l * DBC_W * D_INNER, D_INNER,
            dbc, dbcbf, DBC_W, D_INNER);

        // delta = dbc[:, :32] @ dt_proj_w^T   (8192 x 1024, K=32)
        gemm_nt_wmma<<<dim3(D_INNER / BN, NTOK / BM), 256, 0, stream>>>(
            dbcbf, DBC_W, dtwbf + (size_t)l * D_INNER * DT_RANK, DT_RANK,
            delta, (__bf16*)nullptr, D_INNER, DT_RANK);

        // delta = softplus(delta + dt_proj_b)
        softplus_bias_kernel<<<(NTOK * D_INNER) / 256, 256, 0, stream>>>(
            delta, dtb + (size_t)l * D_INNER);

        // y = (scan(...) + D*u) * silu(z), written directly as bf16
        scan_kernel<<<BATCH * (D_INNER / 256), 256, 0, stream>>>(
            delta, dbc, xc, xz,
            A_log + (size_t)l * D_INNER * D_STATE, Dp + (size_t)l * D_INNER, ybf);

        // h = y @ out_proj_w^T            (8192 x 512, K=1024)  [+ bf16 copy]
        gemm_nt_wmma<<<dim3(D_MODEL / BN, NTOK / BM), 256, 0, stream>>>(
            ybf, D_INNER, opwbf + (size_t)l * D_MODEL * D_INNER, D_INNER,
            h, hbf, D_MODEL, D_INNER);
    }

    layernorm_mask_kernel<<<NTOK, 128, 0, stream>>>(h, nw, nb, mask, (float*)d_out);
}
